// sa_conv2d_59459527246335
// MI455X (gfx1250) — matrix operand order repaired and checked by reading, not compiled, not run on hardware
//
#include <hip/hip_runtime.h>
#include <hip/hip_bf16.h>

// ---------------------------------------------------------------------------
// Problem constants:  x:(8,128,128,256)  W*: (256,256)  rel:(128,7)  K=7
// ---------------------------------------------------------------------------
constexpr int Bc   = 8;
constexpr int Hc   = 128;
constexpr int Wc   = 128;
constexpr int Cc   = 256;   // input channels (GEMM K)
constexpr int Fc   = 256;   // projection dim
constexpr int Kw   = 7;
constexpr int PADc = 3;
constexpr int RS   = 128;
constexpr int Np   = Bc * Hc * Wc;      // 131072 pixels

typedef float        v8f   __attribute__((ext_vector_type(8)));
typedef unsigned int v4u   __attribute__((ext_vector_type(4)));
typedef unsigned int v8u   __attribute__((ext_vector_type(8)));
typedef __bf16       v16bf __attribute__((ext_vector_type(16)));

__device__ __forceinline__ unsigned int pack_bf16(__bf16 lo, __bf16 hi) {
    return ((unsigned int)__builtin_bit_cast(unsigned short, hi) << 16) |
            (unsigned int)__builtin_bit_cast(unsigned short, lo);
}

// Async global->LDS copy, 16B per active lane (GLOBAL_LOAD_ASYNC_TO_LDS_B128,
// GV mode). lds_off is the wave-relative LDS byte address = low 32 bits of a
// generic pointer to __shared__ (flat LDS aperture truncates to addr[31:0]).
// Tracked by ASYNCcnt; pair with async_wait0() before the barrier.
__device__ __forceinline__ void async_copy16(const void* lds_ptr, const void* gptr) {
    const unsigned lds_off = (unsigned)(unsigned long long)lds_ptr;
    asm volatile("global_load_async_to_lds_b128 %0, %1, off"
                 :: "v"(lds_off), "v"((unsigned long long)gptr)
                 : "memory");
}
__device__ __forceinline__ void async_wait0() {
    asm volatile("s_wait_asynccnt 0x0" ::: "memory");
}

// ===========================================================================
// Kernel 0: one-shot weight decomposition W = Wh + Wl (bf16 hi/lo planes),
// stored transposed [m][f][c/2] with consecutive-K bf16 pairs packed per
// dword -- contiguous k order (k = 16h + s) this code uses for A and B (B 32x16, VGPR g
// holds K={2g,2g+1} for lanes 0-15 and K={16+2g,16+2g+1} for 16-31), so a
// lane's whole B fragment for one k-step is 8 contiguous dwords.
// ===========================================================================
__global__ __launch_bounds__(256)
void prep_weights(const float* __restrict__ Wq, const float* __restrict__ Wk,
                  const float* __restrict__ Wv,
                  unsigned int* __restrict__ Wh, unsigned int* __restrict__ Wl)
{
    const int idx = blockIdx.x * 256 + threadIdx.x;       // [m][f][c2]
    if (idx >= 3 * Fc * (Cc / 2)) return;
    const int c2 = idx % (Cc / 2);
    const int f  = (idx / (Cc / 2)) % Fc;
    const int m  = idx / ((Cc / 2) * Fc);
    const float* Wm = (m == 0) ? Wq : (m == 1) ? Wk : Wv;

    const float w0 = Wm[(2 * c2 + 0) * Fc + f];
    const float w1 = Wm[(2 * c2 + 1) * Fc + f];
    const __bf16 h0 = (__bf16)w0, h1 = (__bf16)w1;
    const __bf16 l0 = (__bf16)(w0 - (float)h0), l1 = (__bf16)(w1 - (float)h1);
    Wh[idx] = pack_bf16(h0, h1);
    Wl[idx] = pack_bf16(l0, l1);
}

// ===========================================================================
// Kernel 1: fused QKV projection via bf16x3 split-precision WMMA.
//   q|k|v = X*W with X = Xh+Xl, W = Wh+Wl; acc += Ah*Bh + Ah*Bl + Al*Bh
//   (fp32 accumulate in V_WMMA_F32_16X16X32_BF16, ~fp32 accuracy).
// Block: 256 threads / 8 waves; 16-row X tile decomposed into pair-packed
// hi/lo LDS planes (stride 132 dwords for bank spread). 48 output tiles
// (3 matrices x 16 col-tiles) split 6 per wave; 8 k-steps of 32 each.
// ===========================================================================
__global__ __launch_bounds__(256)
void qkv_wmma_gemm(const float* __restrict__ x,
                   const unsigned int* __restrict__ Wh,
                   const unsigned int* __restrict__ Wl,
                   float* __restrict__ q, float* __restrict__ k,
                   float* __restrict__ v)
{
    constexpr int PR = Cc / 2;            // 128 packed pairs per row
    constexpr int TS = 132;               // padded LDS pair stride (dwords)
    __shared__ unsigned int xh_p[16 * TS];
    __shared__ unsigned int xl_p[16 * TS];

    const int tid = threadIdx.x;
    const int n0  = blockIdx.x * 16;

    // Stage + decompose the 16x256 X tile (float4 in, 2 packed dwords out).
    for (int idx = tid; idx < 16 * (Cc / 4); idx += 256) {
        const int r  = idx / (Cc / 4);
        const int q4 = idx % (Cc / 4);
        const float4 xv = *(const float4*)&x[(n0 + r) * Cc + q4 * 4];
        const __bf16 h0 = (__bf16)xv.x, h1 = (__bf16)xv.y,
                     h2 = (__bf16)xv.z, h3 = (__bf16)xv.w;
        xh_p[r * TS + q4 * 2 + 0] = pack_bf16(h0, h1);
        xh_p[r * TS + q4 * 2 + 1] = pack_bf16(h2, h3);
        xl_p[r * TS + q4 * 2 + 0] = pack_bf16((__bf16)(xv.x - (float)h0),
                                              (__bf16)(xv.y - (float)h1));
        xl_p[r * TS + q4 * 2 + 1] = pack_bf16((__bf16)(xv.z - (float)h2),
                                              (__bf16)(xv.w - (float)h3));
    }
    __syncthreads();

    const int wave = tid >> 5;
    const int lane = tid & 31;
    const int hv   = lane >> 4;           // half-wave
    const int l    = lane & 15;

    float* Outs[3] = {q, k, v};

    #pragma unroll 1
    for (int t = 0; t < 6; ++t) {
        const int tile = wave * 6 + t;            // 0..47
        const int m    = tile >> 4;               // matrix select
        const int f0   = (tile & 15) * 16;        // output column tile
        const unsigned int* wh_row = Wh + (size_t)(m * Fc + f0 + l) * PR;
        const unsigned int* wl_row = Wl + (size_t)(m * Fc + f0 + l) * PR;
        const unsigned int* ah_row = &xh_p[l * TS];
        const unsigned int* al_row = &xl_p[l * TS];

        v8f acc = {};
        #pragma unroll 2
        for (int kc2 = 0; kc2 < PR; kc2 += 16) {  // k-step of 32 channels
            // A frag (16-bit A 16x32): pair runs pa..pa+3 and pa+4..pa+7.
            const int pa = kc2 + hv * 8;
            const v4u ah0 = *(const v4u*)&ah_row[pa];
            const v4u ah1 = *(const v4u*)&ah_row[pa + 4];
            const v4u al0 = *(const v4u*)&al_row[pa];
            const v4u al1 = *(const v4u*)&al_row[pa + 4];
            const v16bf Ah = __builtin_bit_cast(
                v16bf, __builtin_shufflevector(ah0, ah1, 0,1,2,3,4,5,6,7));
            const v16bf Al = __builtin_bit_cast(
                v16bf, __builtin_shufflevector(al0, al1, 0,1,2,3,4,5,6,7));

            const int pb = kc2 + hv * 8;          // B: 8 contiguous dwords
            const v16bf Bh = __builtin_bit_cast(v16bf, *(const v8u*)&wh_row[pb]);
            const v16bf Bl = __builtin_bit_cast(v16bf, *(const v8u*)&wl_row[pb]);

            acc = __builtin_amdgcn_wmma_f32_16x16x32_bf16(
                      false, Ah, false, Bh, (short)0, acc, false, false);
            acc = __builtin_amdgcn_wmma_f32_16x16x32_bf16(
                      false, Ah, false, Bl, (short)0, acc, false, false);
            acc = __builtin_amdgcn_wmma_f32_16x16x32_bf16(
                      false, Al, false, Bh, (short)0, acc, false, false);
        }

        float* Om = Outs[m];
        #pragma unroll
        for (int r = 0; r < 8; ++r)               // D: VGPR r -> M = r + hv*8
            Om[(n0 + r + hv * 8) * Fc + f0 + l] = acc[r];
    }
}

// ===========================================================================
// Kernel 2: windowed attention. Bandwidth-bound; LDS halo tiling with
// ASYNC global->LDS staging (no VGPR round-trip); zero-pad semantics match
// the reference (OOB score = 0 + rel terms, inside softmax; OOB v adds 0).
// ===========================================================================
constexpr int TP     = 32;            // pixels per block
constexpr int CH     = 64;            // channel chunk
constexpr int HALO_W = TP + Kw - 1;   // 38
constexpr int HST    = 68;            // padded halo channel stride

__global__ __launch_bounds__(256)
void window_attn(const float* __restrict__ q,
                 const float* __restrict__ k,
                 const float* __restrict__ v,
                 const float* __restrict__ rel_x,
                 const float* __restrict__ rel_y,
                 const float* __restrict__ bias,
                 float* __restrict__ out)
{
    __shared__ float q_tile[TP * HST];
    __shared__ float halo[Kw * HALO_W * HST];
    __shared__ float sc[TP * Kw * Kw];
    __shared__ float qx_s[TP * Kw];
    __shared__ float qy_s[TP * Kw];

    const int tid = threadIdx.x;
    const int w0  = blockIdx.x * TP;
    const int h   = blockIdx.y;
    const int b   = blockIdx.z;
    const int nbase = (b * Hc + h) * Wc + w0;

    // ---- relative-position terms: qx[p][j] (rel_x), qy[p][i] (rel_y) ----
    for (int i = tid; i < TP * Kw; i += 256) { qx_s[i] = 0.f; qy_s[i] = 0.f; }
    __syncthreads();
    {
        const int p = tid >> 3, s = tid & 7;
        const float* qrow = q + (size_t)(nbase + p) * Fc;
        float ax[Kw] = {}, ay[Kw] = {};
        for (int c = s * 16; c < s * 16 + 16; ++c) {
            const float q0 = qrow[c];
            const float q1 = qrow[RS + c];
            #pragma unroll
            for (int j = 0; j < Kw; ++j) {
                ax[j] += q0 * rel_x[c * Kw + j];
                ay[j] += q1 * rel_y[c * Kw + j];
            }
        }
        #pragma unroll
        for (int j = 0; j < Kw; ++j) {
            atomicAdd(&qx_s[p * Kw + j], ax[j]);
            atomicAdd(&qy_s[p * Kw + j], ay[j]);
        }
    }
    __syncthreads();

    // ---- scores ----
    for (int cc = 0; cc < Cc / CH; ++cc) {
        const int cbase = cc * CH;
        // q tile chunk: pure async copy (always in bounds)
        for (int idx = tid; idx < TP * (CH / 4); idx += 256) {
            const int p  = idx / (CH / 4);
            const int c4 = (idx % (CH / 4)) * 4;
            async_copy16(&q_tile[p * HST + c4],
                         &q[(size_t)(nbase + p) * Fc + cbase + c4]);
        }
        // k halo chunk: async copy in-bounds, ds-store zeros out-of-bounds
        for (int idx = tid; idx < Kw * HALO_W * (CH / 4); idx += 256) {
            int t = idx;
            const int c4 = (t % (CH / 4)) * 4;  t /= (CH / 4);
            const int xx = t % HALO_W;          t /= HALO_W;
            const int i  = t;
            const int hh = h + i - PADc, ww = w0 + xx - PADc;
            float* dst = &halo[(i * HALO_W + xx) * HST + c4];
            if (hh >= 0 && hh < Hc && ww >= 0 && ww < Wc)
                async_copy16(dst, &k[(size_t)((b * Hc + hh) * Wc + ww) * Fc + cbase + c4]);
            else
                *(float4*)dst = make_float4(0.f, 0.f, 0.f, 0.f);
        }
        async_wait0();
        __syncthreads();

        for (int sidx = tid; sidx < TP * Kw * Kw; sidx += 256) {
            const int p  = sidx / (Kw * Kw);
            const int ij = sidx % (Kw * Kw);
            const int i  = ij / Kw, j = ij % Kw;
            float acc = (cc == 0) ? (qx_s[p * Kw + j] + qy_s[p * Kw + i])
                                  : sc[sidx];
            const float* qp = &q_tile[p * HST];
            const float* kp = &halo[(i * HALO_W + (p + j)) * HST];
            #pragma unroll 8
            for (int c = 0; c < CH; ++c) acc += qp[c] * kp[c];
            sc[sidx] = acc;
        }
        __syncthreads();
    }

    // ---- softmax over 49 ----
    if (tid < TP) {
        float* s = &sc[tid * Kw * Kw];
        float m = s[0];
        #pragma unroll
        for (int ij = 1; ij < Kw * Kw; ++ij) m = fmaxf(m, s[ij]);
        float sum = 0.f;
        #pragma unroll
        for (int ij = 0; ij < Kw * Kw; ++ij) { const float e = expf(s[ij] - m); s[ij] = e; sum += e; }
        const float inv = 1.f / sum;
        #pragma unroll
        for (int ij = 0; ij < Kw * Kw; ++ij) s[ij] *= inv;
    }
    __syncthreads();

    // ---- weighted V sum + bias ----
    for (int cc = 0; cc < Cc / CH; ++cc) {
        const int cbase = cc * CH;
        for (int idx = tid; idx < Kw * HALO_W * (CH / 4); idx += 256) {
            int t = idx;
            const int c4 = (t % (CH / 4)) * 4;  t /= (CH / 4);
            const int xx = t % HALO_W;          t /= HALO_W;
            const int i  = t;
            const int hh = h + i - PADc, ww = w0 + xx - PADc;
            float* dst = &halo[(i * HALO_W + xx) * HST + c4];
            if (hh >= 0 && hh < Hc && ww >= 0 && ww < Wc)
                async_copy16(dst, &v[(size_t)((b * Hc + hh) * Wc + ww) * Fc + cbase + c4]);
            else
                *(float4*)dst = make_float4(0.f, 0.f, 0.f, 0.f);
        }
        async_wait0();
        __syncthreads();

        for (int oidx = tid; oidx < TP * CH; oidx += 256) {
            const int p = oidx / CH, c = oidx % CH;
            float acc = bias[cbase + c];
            const float* ap = &sc[p * Kw * Kw];
            #pragma unroll
            for (int i = 0; i < Kw; ++i) {
                const float* vrow = &halo[(i * HALO_W + p) * HST + c];
                #pragma unroll
                for (int j = 0; j < Kw; ++j)
                    acc += ap[i * Kw + j] * vrow[j * HST];
            }
            out[(size_t)(nbase + p) * Fc + cbase + c] = acc;
        }
        __syncthreads();
    }
}

// ===========================================================================
// Launch. d_ws layout: q|k|v (3 * 131072*256 f32 = 384 MB) then Wh|Wl planes
// (2 * 3*256*128 dwords = 768 KB). All on `stream`; no alloc/sync here.
// ===========================================================================
extern "C" void kernel_launch(void* const* d_in, const int* in_sizes, int n_in,
                              void* d_out, int out_size, void* d_ws, size_t ws_size,
                              hipStream_t stream) {
    const float* x     = (const float*)d_in[0];
    const float* Wq    = (const float*)d_in[1];
    const float* Wk    = (const float*)d_in[2];
    const float* Wv    = (const float*)d_in[3];
    const float* rel_x = (const float*)d_in[4];
    const float* rel_y = (const float*)d_in[5];
    const float* bias  = (const float*)d_in[6];
    float* out = (float*)d_out;

    const size_t NF = (size_t)Np * Fc;
    float* q = (float*)d_ws;
    float* k = q + NF;
    float* v = k + NF;
    unsigned int* Wh = (unsigned int*)(v + NF);
    unsigned int* Wl = Wh + (size_t)3 * Fc * (Cc / 2);

    const int nprep = 3 * Fc * (Cc / 2);
    prep_weights<<<(nprep + 255) / 256, 256, 0, stream>>>(Wq, Wk, Wv, Wh, Wl);
    qkv_wmma_gemm<<<Np / 16, 256, 0, stream>>>(x, Wh, Wl, q, k, v);
    window_attn<<<dim3(Wc / TP, Hc, Bc), 256, 0, stream>>>(q, k, v, rel_x, rel_y, bias, out);
}
